// LM_QAGNN_36249523978271
// MI455X (gfx1250) — matrix operand order repaired and checked by reading, not compiled, not run on hardware
//
#include <hip/hip_runtime.h>
#include <hip/hip_bf16.h>
#include <math.h>

typedef __attribute__((ext_vector_type(16))) _Float16 v16h;
typedef __attribute__((ext_vector_type(8)))  _Float16 v8h;
typedef __attribute__((ext_vector_type(8)))  float    v8f;

constexpr int B_  = 16;
constexpr int L_  = 512;
constexpr int DS_ = 1024;
constexpr int N_  = 200;
constexpr int E_  = 1600;
constexpr int H_  = 256;
constexpr int NP_ = 208;   // N padded to multiple of 16

// ---------------------------------------------------------------------------
// fp32 (R,C) -> f16 (C,R) transpose-convert, batched over blockIdx.z
// ---------------------------------------------------------------------------
__global__ void transpose_f32_to_f16(const float* __restrict__ in,
                                     _Float16* __restrict__ out,
                                     int R, int C, long inBatch, long outBatch) {
  __shared__ float tile[32][33];
  const float* src = in + (long)blockIdx.z * inBatch;
  _Float16* dst = out + (long)blockIdx.z * outBatch;
  int c0 = blockIdx.x * 32, r0 = blockIdx.y * 32;
  for (int i = threadIdx.y; i < 32; i += 8) {
    int r = r0 + i, c = c0 + threadIdx.x;
    tile[i][threadIdx.x] = (r < R && c < C) ? src[(long)r * C + c] : 0.f;
  }
  __syncthreads();
  for (int i = threadIdx.y; i < 32; i += 8) {
    int c = c0 + i, r = r0 + threadIdx.x;
    if (c < C && r < R) dst[(long)c * R + r] = (_Float16)tile[threadIdx.x][i];
  }
}

__global__ void convert_f32_to_f16(const float* __restrict__ in,
                                   _Float16* __restrict__ out, long n) {
  long i = (long)blockIdx.x * blockDim.x + threadIdx.x;
  if (i < n) out[i] = (_Float16)in[i];
}

__global__ void zero_f32(float* __restrict__ p, long n) {
  long i = (long)blockIdx.x * blockDim.x + threadIdx.x;
  if (i < n) p[i] = 0.f;
}

// ---------------------------------------------------------------------------
// so_mask = em | shift_left(em) | shift_right(em); f16 masks + 1/max(cnt,1)
// one block per (b, n) row; rows n>=N are zero padding
// ---------------------------------------------------------------------------
__global__ void prep_masks(const int* __restrict__ em,
                           _Float16* __restrict__ soH, _Float16* __restrict__ emH,
                           float* __restrict__ invSo, float* __restrict__ invEm) {
  int bn = blockIdx.x;                 // b*NP + n
  int b = bn / NP_, n = bn % NP_;
  __shared__ int cSo, cEm;
  if (threadIdx.x == 0) { cSo = 0; cEm = 0; }
  __syncthreads();
  long outBase = (long)bn * L_;
  if (n < N_) {
    const int* row = em + ((long)b * N_ + n) * L_;
    int mySo = 0, myEm = 0;
    for (int l = threadIdx.x; l < L_; l += blockDim.x) {
      int c  = row[l];
      int lm = (l > 0)      ? row[l - 1] : 0;
      int lp = (l < L_ - 1) ? row[l + 1] : 0;
      int s = c | lm | lp;
      soH[outBase + l] = (_Float16)(float)s;
      emH[outBase + l] = (_Float16)(float)c;
      mySo += s; myEm += c;
    }
    atomicAdd(&cSo, mySo);
    atomicAdd(&cEm, myEm);
  } else {
    for (int l = threadIdx.x; l < L_; l += blockDim.x) {
      soH[outBase + l] = (_Float16)0.f;
      emH[outBase + l] = (_Float16)0.f;
    }
  }
  __syncthreads();
  if (threadIdx.x == 0) {
    invSo[bn] = 1.f / fmaxf((float)cSo, 1.f);
    invEm[bn] = 1.f / fmaxf((float)cEm, 1.f);
  }
}

// ---------------------------------------------------------------------------
// Generic batched WMMA GEMM:  C[z] = A[z] (MxK, row-major f16)
//                                    x Bt[z]^T (Bt is N-major: Nn x K, f16)
// optional per-row scale, per-col bias; f32 and/or f16 outputs.
// block = 256 threads (8 waves); wave w -> N-tile w; grid (M/16, Nn/128, nbatch)
// ---------------------------------------------------------------------------
__global__ void __launch_bounds__(256)
gemm_wmma_f16(const _Float16* __restrict__ A, long sA,
              const _Float16* __restrict__ Bt, long sB,
              float* __restrict__ C32, _Float16* __restrict__ C16, long sC,
              const float* __restrict__ rowScale, int rss,
              const float* __restrict__ bias,
              int Nn, int K) {
  int lane = threadIdx.x & 31;
  int wave = threadIdx.x >> 5;
  int z = blockIdx.z;
  int m0 = blockIdx.x * 16;
  int n0 = blockIdx.y * 128 + wave * 16;
  int hl = lane >> 4;                       // half-wave select
  const _Float16* Ab = A + (long)z * sA;
  const _Float16* Bb = Bt + (long)z * sB;
  const _Float16* aRow = Ab + (long)(m0 + (lane & 15)) * K + hl * 16;
  const _Float16* bRow = Bb + (long)(n0 + (lane & 15)) * K + hl * 16;
  v8f acc = {0.f, 0.f, 0.f, 0.f, 0.f, 0.f, 0.f, 0.f};
  for (int k0 = 0; k0 < K; k0 += 32) {
    __builtin_prefetch(bRow + k0 + 256, 0, 1);
    v8h alo = *(const v8h*)(aRow + k0);
    v8h ahi = *(const v8h*)(aRow + k0 + 8);
    v16h a = __builtin_shufflevector(alo, ahi, 0,1,2,3,4,5,6,7,8,9,10,11,12,13,14,15);
    v16h b = *(const v16h*)(bRow + k0);
    acc = __builtin_amdgcn_wmma_f32_16x16x32_f16(false, a, false, b,
                                                 (short)0, acc, false, false);
  }
  int colg = n0 + (lane & 15);
  float bv = bias ? bias[colg] : 0.f;
#pragma unroll
  for (int i = 0; i < 8; ++i) {
    int rg = m0 + i + hl * 8;
    float v = acc[i];
    if (rowScale) v *= rowScale[(long)z * rss + rg];
    v += bv;
    long idx = (long)z * sC + (long)rg * Nn + colg;
    if (C32) C32[idx] = v;
    if (C16) C16[idx] = (_Float16)v;
  }
}

// ---------------------------------------------------------------------------
// Per-edge: r_weight = 0.5*(Q[cs]+Q[co]); msg = r_weight * x[src];
// scatter-add to sums[dst], count edges per dst.
// one block per (b,e); 256 threads = H
// ---------------------------------------------------------------------------
__global__ void edge_message(const int* __restrict__ edge_index,
                             const int* __restrict__ concept_ids,
                             const float* __restrict__ Q,
                             const float* __restrict__ X,
                             float* __restrict__ sums, float* __restrict__ cnt) {
  int be = blockIdx.x;
  int b = be / E_, e = be % E_;
  int src = edge_index[((long)b * 2 + 0) * E_ + e];
  int dst = edge_index[((long)b * 2 + 1) * E_ + e];
  int cs = concept_ids[b * N_ + src];
  int co = concept_ids[b * N_ + dst];
  int h = threadIdx.x;
  float rw = 0.5f * (Q[((long)b * NP_ + cs) * H_ + h] +
                     Q[((long)b * NP_ + co) * H_ + h]);
  float msg = rw * X[((long)b * NP_ + src) * H_ + h];
  atomicAdd(&sums[((long)b * N_ + dst) * H_ + h], msg);
  if (h == 0) atomicAdd(&cnt[b * N_ + dst], 1.f);
}

// node_out = sums/max(cnt,1) + x@root + message_bias ; msg = relu(node_out+gru_bias)
__global__ void node_out_kernel(const float* __restrict__ sums,
                                const float* __restrict__ cnt,
                                const float* __restrict__ xr,
                                const float* __restrict__ message_bias,
                                const float* __restrict__ gru_bias,
                                float* __restrict__ node_out,
                                _Float16* __restrict__ msgH) {
  int bn = blockIdx.x;                 // b*N + n
  int b = bn / N_, n = bn % N_;
  int h = threadIdx.x;
  float c = fmaxf(cnt[bn], 1.f);
  long idx = (long)bn * H_ + h;
  float v = sums[idx] / c + xr[((long)b * NP_ + n) * H_ + h] + message_bias[h];
  node_out[idx] = v;
  msgH[((long)n * B_ + b) * H_ + h] = (_Float16)fmaxf(v + gru_bias[h], 0.f);
}

__global__ void h0_kernel(const float* __restrict__ node_out, float* __restrict__ h0) {
  int b = blockIdx.x, h = threadIdx.x;
  float m = -3.4e38f;
  for (int n = 0; n < N_; ++n)
    m = fmaxf(m, node_out[((long)b * N_ + n) * H_ + h]);
  h0[b * H_ + h] = m;
}

// ---------------------------------------------------------------------------
// GRU scan. blockIdx.x = direction (0 fwd, 1 bwd). 256 threads = 8 waves.
// h kept in LDS (f16 for WMMA-A + f32 master). Wave w owns gate columns
// [32w, 32w+32): 6 accumulators = {r,z,n} x {2 col-tiles}, so gate math stays
// inside the wave that produced gh. Writes out[b,t, dir*H + j].
// ---------------------------------------------------------------------------
__global__ void __launch_bounds__(256)
gru_scan(const float* __restrict__ gx,        // (2, N, B, 3H) f32
         const _Float16* __restrict__ WhhH,   // (2, 3H, H) f16 (N-major for B-op)
         const float* __restrict__ bhh,       // (2, 3H)
         const float* __restrict__ h0,        // (B, H)
         float* __restrict__ out) {           // (B, N, 2H)
  int dir = blockIdx.x;
  const float* gxd = gx + (long)dir * N_ * B_ * 3 * H_;
  const _Float16* Wd = WhhH + (long)dir * 3 * H_ * H_;
  const float* bhhd = bhh + dir * 3 * H_;

  __shared__ _Float16 h16[B_ * H_];   // 8 KB
  __shared__ float    h32[B_ * H_];   // 16 KB
  __shared__ float    bhhs[3 * H_];   // 3 KB

  for (int i = threadIdx.x; i < B_ * H_; i += 256) {
    float v = h0[i];
    h32[i] = v;
    h16[i] = (_Float16)v;
  }
  for (int i = threadIdx.x; i < 3 * H_; i += 256) bhhs[i] = bhhd[i];
  __syncthreads();

  int lane = threadIdx.x & 31;
  int wave = threadIdx.x >> 5;
  int hl = lane >> 4;
  // B-fragment row pointers: gate g, col-tile p -> n = g*H + (2w+p)*16 + (lane&15)
  const _Float16* bRow[3][2];
#pragma unroll
  for (int g = 0; g < 3; ++g)
#pragma unroll
    for (int p = 0; p < 2; ++p)
      bRow[g][p] = Wd + (long)(g * H_ + (2 * wave + p) * 16 + (lane & 15)) * H_ + hl * 16;
  const _Float16* aBase = &h16[(lane & 15) * H_ + hl * 16];

  for (int step = 0; step < N_; ++step) {
    int t = (dir == 0) ? step : (N_ - 1 - step);
    v8f acc[3][2];
#pragma unroll
    for (int g = 0; g < 3; ++g)
#pragma unroll
      for (int p = 0; p < 2; ++p)
        acc[g][p] = (v8f){0.f, 0.f, 0.f, 0.f, 0.f, 0.f, 0.f, 0.f};

    for (int k0 = 0; k0 < H_; k0 += 32) {
      v8h alo = *(const v8h*)(aBase + k0);
      v8h ahi = *(const v8h*)(aBase + k0 + 8);
      v16h a = __builtin_shufflevector(alo, ahi, 0,1,2,3,4,5,6,7,8,9,10,11,12,13,14,15);
#pragma unroll
      for (int g = 0; g < 3; ++g)
#pragma unroll
        for (int p = 0; p < 2; ++p) {
          v16h bb = *(const v16h*)(bRow[g][p] + k0);
          acc[g][p] = __builtin_amdgcn_wmma_f32_16x16x32_f16(false, a, false, bb,
                                                             (short)0, acc[g][p],
                                                             false, false);
        }
    }
    __syncthreads();   // everyone done reading h16 this step

#pragma unroll
    for (int p = 0; p < 2; ++p) {
      int j = (2 * wave + p) * 16 + (lane & 15);
#pragma unroll
      for (int i = 0; i < 8; ++i) {
        int b = i + hl * 8;
        float hr = acc[0][p][i] + bhhs[j];
        float hz = acc[1][p][i] + bhhs[H_ + j];
        float hn = acc[2][p][i] + bhhs[2 * H_ + j];
        long gbase = ((long)t * B_ + b) * (3 * H_);
        float xr_ = gxd[gbase + j];
        float xz_ = gxd[gbase + H_ + j];
        float xn_ = gxd[gbase + 2 * H_ + j];
        float r  = 1.f / (1.f + expf(-(xr_ + hr)));
        float zt = 1.f / (1.f + expf(-(xz_ + hz)));
        float nn = tanhf(xn_ + r * hn);
        float hp = h32[b * H_ + j];
        float hnew = (1.f - zt) * nn + zt * hp;
        out[((long)b * N_ + t) * (2 * H_) + dir * H_ + j] = hnew;
        h32[b * H_ + j] = hnew;
        h16[b * H_ + j] = (_Float16)hnew;
      }
    }
    __syncthreads();   // h updated before next step's A-fragment reads
  }
}

// ---------------------------------------------------------------------------
// Host side
// ---------------------------------------------------------------------------
extern "C" void kernel_launch(void* const* d_in, const int* in_sizes, int n_in,
                              void* d_out, int out_size, void* d_ws, size_t ws_size,
                              hipStream_t stream) {
  const float* batch_embedding = (const float*)d_in[0];
  const int*   entity_mask     = (const int*)d_in[1];
  const int*   concept_ids     = (const int*)d_in[2];
  const int*   edge_index      = (const int*)d_in[3];
  const float* W_node          = (const float*)d_in[4];
  const float* W_edge          = (const float*)d_in[5];
  const float* l_weight        = (const float*)d_in[6];
  const float* root            = (const float*)d_in[7];
  const float* message_bias    = (const float*)d_in[8];
  const float* gru_Wih         = (const float*)d_in[9];
  const float* gru_Whh         = (const float*)d_in[10];
  const float* gru_bih         = (const float*)d_in[11];
  const float* gru_bhh         = (const float*)d_in[12];
  const float* gru_bias        = (const float*)d_in[13];
  float* out = (float*)d_out;

  char* ws = (char*)d_ws;
  size_t off = 0;
  auto alloc = [&](size_t bytes) -> char* {
    char* p = ws + off;
    off += (bytes + 255) & ~(size_t)255;
    return p;
  };
  _Float16* embT    = (_Float16*)alloc((size_t)B_ * DS_ * L_ * 2);   // (B, DS, L)
  _Float16* soH     = (_Float16*)alloc((size_t)B_ * NP_ * L_ * 2);   // (B, NP, L)
  _Float16* emH     = (_Float16*)alloc((size_t)B_ * NP_ * L_ * 2);
  float*    invSo   = (float*)alloc((size_t)B_ * NP_ * 4);
  float*    invEm   = (float*)alloc((size_t)B_ * NP_ * 4);
  _Float16* soMeanH = (_Float16*)alloc((size_t)B_ * NP_ * DS_ * 2);  // (B, NP, DS)
  _Float16* nodeH   = (_Float16*)alloc((size_t)B_ * NP_ * DS_ * 2);
  _Float16* WeT     = (_Float16*)alloc((size_t)H_ * DS_ * 2);        // (H, DS)
  _Float16* WnT     = (_Float16*)alloc((size_t)H_ * DS_ * 2);
  _Float16* lT      = (_Float16*)alloc((size_t)H_ * H_ * 2);
  _Float16* rT      = (_Float16*)alloc((size_t)H_ * H_ * 2);
  _Float16* WhhH    = (_Float16*)alloc((size_t)2 * 3 * H_ * H_ * 2);
  _Float16* WihH    = (_Float16*)alloc((size_t)2 * 3 * H_ * H_ * 2);
  _Float16* PH      = (_Float16*)alloc((size_t)B_ * NP_ * H_ * 2);
  float*    Qf      = (float*)alloc((size_t)B_ * NP_ * H_ * 4);
  float*    X32     = (float*)alloc((size_t)B_ * NP_ * H_ * 4);
  _Float16* XH      = (_Float16*)alloc((size_t)B_ * NP_ * H_ * 2);
  float*    XR      = (float*)alloc((size_t)B_ * NP_ * H_ * 4);
  float*    sums    = (float*)alloc((size_t)B_ * N_ * H_ * 4);
  float*    cntb    = (float*)alloc((size_t)B_ * N_ * 4);
  float*    nodeOut = (float*)alloc((size_t)B_ * N_ * H_ * 4);
  _Float16* msgH    = (_Float16*)alloc((size_t)N_ * B_ * H_ * 2);    // (N, B, H)
  float*    h0b     = (float*)alloc((size_t)B_ * H_ * 4);
  float*    gxB     = (float*)alloc((size_t)2 * N_ * B_ * 3 * H_ * 4);

  dim3 tb(32, 8);
  // embT[b] = emb[b]^T  (R=L, C=DS)
  transpose_f32_to_f16<<<dim3(DS_ / 32, L_ / 32, B_), tb, 0, stream>>>(
      batch_embedding, embT, L_, DS_, (long)L_ * DS_, (long)DS_ * L_);
  // W_edge^T (R=DS, C=H), W_node^T, l_weight^T, root^T
  transpose_f32_to_f16<<<dim3(H_ / 32, DS_ / 32, 1), tb, 0, stream>>>(
      W_edge, WeT, DS_, H_, 0, 0);
  transpose_f32_to_f16<<<dim3(H_ / 32, DS_ / 32, 1), tb, 0, stream>>>(
      W_node, WnT, DS_, H_, 0, 0);
  transpose_f32_to_f16<<<dim3(H_ / 32, H_ / 32, 1), tb, 0, stream>>>(
      l_weight, lT, H_, H_, 0, 0);
  transpose_f32_to_f16<<<dim3(H_ / 32, H_ / 32, 1), tb, 0, stream>>>(
      root, rT, H_, H_, 0, 0);
  { long n = (long)2 * 3 * H_ * H_;
    convert_f32_to_f16<<<(n + 255) / 256, 256, 0, stream>>>(gru_Whh, WhhH, n);
    convert_f32_to_f16<<<(n + 255) / 256, 256, 0, stream>>>(gru_Wih, WihH, n); }

  prep_masks<<<B_ * NP_, 256, 0, stream>>>(entity_mask, soH, emH, invSo, invEm);

  // so_mean = so @ embT^T / cnt  (M=NP, Nn=DS, K=L), f16 out
  gemm_wmma_f16<<<dim3(NP_ / 16, DS_ / 128, B_), 256, 0, stream>>>(
      soH, (long)NP_ * L_, embT, (long)DS_ * L_,
      nullptr, soMeanH, (long)NP_ * DS_, invSo, NP_, nullptr, DS_, L_);
  // node_emb = em @ embT^T / cnt
  gemm_wmma_f16<<<dim3(NP_ / 16, DS_ / 128, B_), 256, 0, stream>>>(
      emH, (long)NP_ * L_, embT, (long)DS_ * L_,
      nullptr, nodeH, (long)NP_ * DS_, invEm, NP_, nullptr, DS_, L_);
  // P = so_mean @ W_edge   (M=NP, Nn=H, K=DS), f16 out
  gemm_wmma_f16<<<dim3(NP_ / 16, H_ / 128, B_), 256, 0, stream>>>(
      soMeanH, (long)NP_ * DS_, WeT, 0,
      nullptr, PH, (long)NP_ * H_, nullptr, 0, nullptr, H_, DS_);
  // Q = P @ l_weight       (K=H), f32 out
  gemm_wmma_f16<<<dim3(NP_ / 16, H_ / 128, B_), 256, 0, stream>>>(
      PH, (long)NP_ * H_, lT, 0,
      Qf, nullptr, (long)NP_ * H_, nullptr, 0, nullptr, H_, H_);
  // x = node_emb @ W_node  (K=DS), f32 + f16 out
  gemm_wmma_f16<<<dim3(NP_ / 16, H_ / 128, B_), 256, 0, stream>>>(
      nodeH, (long)NP_ * DS_, WnT, 0,
      X32, XH, (long)NP_ * H_, nullptr, 0, nullptr, H_, DS_);
  // xr = x @ root          (K=H), f32 out
  gemm_wmma_f16<<<dim3(NP_ / 16, H_ / 128, B_), 256, 0, stream>>>(
      XH, (long)NP_ * H_, rT, 0,
      XR, nullptr, (long)NP_ * H_, nullptr, 0, nullptr, H_, H_);

  { long n = (long)B_ * N_ * H_;
    zero_f32<<<(n + 255) / 256, 256, 0, stream>>>(sums, n); }
  { long n = (long)B_ * N_;
    zero_f32<<<(n + 255) / 256, 256, 0, stream>>>(cntb, n); }

  edge_message<<<B_ * E_, H_, 0, stream>>>(edge_index, concept_ids, Qf, X32, sums, cntb);
  node_out_kernel<<<B_ * N_, H_, 0, stream>>>(sums, cntb, XR, message_bias,
                                              gru_bias, nodeOut, msgH);
  h0_kernel<<<B_, H_, 0, stream>>>(nodeOut, h0b);

  // gx[d] = msg @ Wih[d]^T + bih[d]  (batched over t: M=16, Nn=3H, K=H)
  gemm_wmma_f16<<<dim3(1, (3 * H_) / 128, N_), 256, 0, stream>>>(
      msgH, (long)B_ * H_, WihH, 0,
      gxB, nullptr, (long)B_ * 3 * H_, nullptr, 0, gru_bih, 3 * H_, H_);
  gemm_wmma_f16<<<dim3(1, (3 * H_) / 128, N_), 256, 0, stream>>>(
      msgH, (long)B_ * H_, WihH + (long)3 * H_ * H_, 0,
      gxB + (long)N_ * B_ * 3 * H_, nullptr, (long)B_ * 3 * H_,
      nullptr, 0, gru_bih + 3 * H_, 3 * H_, H_);

  gru_scan<<<2, 256, 0, stream>>>(gxB, WhhH, gru_bhh, h0b, out);
}
